// RTSlowFastRNN_14946486190633
// MI455X (gfx1250) — matrix operand order repaired and checked by reading, compiled, not run on hardware
//
#include <hip/hip_runtime.h>
#include <hip/hip_bf16.h>

typedef __attribute__((ext_vector_type(16))) _Float16 v16h;
typedef __attribute__((ext_vector_type(8)))  _Float16 v8h;
typedef __attribute__((ext_vector_type(8)))  float    v8f;

#define B_DIM   128
#define T_DIM   512
#define HS      128
#define HF      64
#define KSTEPS  15
#define NBT     (B_DIM * T_DIM)           // 65536 rows for the fast stage
#define DELTA_T 0.1f                      // 1.5 / 15

__device__ __forceinline__ float sigmf_(float v) { return 1.0f / (1.0f + __expf(-v)); }

// sum across the 16 lanes of each half-wave group (xor masks stay in-group)
__device__ __forceinline__ float red16_(float v) {
    v += __shfl_xor(v, 1, 32);
    v += __shfl_xor(v, 2, 32);
    v += __shfl_xor(v, 4, 32);
    v += __shfl_xor(v, 8, 32);
    return v;
}

// ---------------------------------------------------------------------------
// WMMA operand fragment builders for v_wmma_f32_16x16x32_f16 (wave32).
// A (16xK, 32-wide K slice), source row-major [m][k], stride ld halves.
// ISA layout: lanes 0-15 hold M=0..15 with K = k0+{0..7, 16..23};
//             lanes 16-31 hold same rows with K = k0+{8..15, 24..31}.
__device__ __forceinline__ v16h load_frag_a(const _Float16* base, int ld, int lane, int k0) {
    int m  = lane & 15;
    int kg = (lane & 16) ? 8 : 0;
    const _Float16* p = base + m * ld + k0 + kg;
    v8h lo = *(const v8h*)(p);
    v8h hi = *(const v8h*)(p + 16);
    v16h a;
#pragma unroll
    for (int i = 0; i < 8; ++i) { a[i] = lo[i]; a[i + 8] = hi[i]; }
    return a;
}

// B (Kx16 tile). Source is a weight matrix W[n][k] row-major (stride ld),
// so B[k][n] = W[n][k] and each lane reads two runs of 8 contiguous K values.
// lanes 0-15 -> col n0+(lane&15), K=k0+{0..7,16..23}; lanes 16-31 -> K=k0+{8..15,24..31}.
__device__ __forceinline__ v16h load_frag_b(const _Float16* wbase, int ld, int n0, int k0, int lane) {
    int n  = n0 + (lane & 15);
    int kg = (lane & 16) ? 8 : 0;
    const _Float16* p = wbase + n * ld + k0 + kg;
    v8h lo = *(const v8h*)(p);
    v8h hi = *(const v8h*)(p + 16);
    v16h b;
#pragma unroll
    for (int i = 0; i < 8; ++i) { b[i] = lo[i]; b[i + 8] = hi[i]; }
    return b;
}

#define WMMA_F16(a, b, c) \
    __builtin_amdgcn_wmma_f32_16x16x32_f16(false, (a), false, (b), (short)0, (c), false, false)

// ---------------------------------------------------------------------------
// Slow GRU: one persistent workgroup (16 waves / 512 threads), sequential scan
// over T=512. Each 16-row batch slab is owned by a *pair* of waves; wave p of
// the pair computes the p-th 64-column half of all three gates (12 WMMA tiles,
// 96 accumulator VGPRs) and updates its half of h. h lives in f32 registers
// (C-layout) plus an f16 mirror in LDS used as next step's WMMA A operand.
// W_hh_s is resident in LDS (f16, 96 KB) -> B-fragments are ds_load_b128,
// software double-buffered so each WMMA overlaps the next fragment's loads.
__global__ __launch_bounds__(512) void slow_gru_kernel(
    const float* __restrict__ x,        // (B,T,4)
    const float* __restrict__ W_hh,     // (384,128) f32
    const float* __restrict__ W_ih,     // (384,4)
    const float* __restrict__ b_ih,     // (384,)
    const float* __restrict__ b_hh,     // (384,)
    const float* __restrict__ W_sc,     // (1,128)
    const float* __restrict__ b_sc,     // (1,)
    _Float16* __restrict__ slow16,      // (B*T,128) f16 out (workspace)
    float* __restrict__ sc_out)         // (B*T,) slice of d_out
{
    __shared__ _Float16 sW[384 * HS];     // 96 KB: W_hh_s f16, [gate_col][k]
    __shared__ _Float16 sH[B_DIM * HS];   // 32 KB: h (f16), row-major [b][k]
    __shared__ float    sWih[384 * 4];    // 6 KB
    __shared__ float    sBih[384];
    __shared__ float    sBhh[384];
    __shared__ float    sX[B_DIM * 4];    // x[:, t, :] staging
    __shared__ float    sSC[B_DIM * 2];   // sc partial per (row, col-half)

    const int tid  = threadIdx.x;
    const int lane = tid & 31;
    const int wave = tid >> 5;            // 0..15
    const int slab = wave >> 1;           // 0..7  (16-row slab)
    const int p    = wave & 1;            // column half of this wave
    const int co   = p * 64;              // first h column owned
    const int l15  = lane & 15;
    const int grp8 = (lane & 16) ? 8 : 0;
    const int rb   = slab * 16;           // first batch row of the slab

    for (int i = tid; i < 384 * HS; i += 512) sW[i] = (_Float16)W_hh[i];
    for (int i = tid; i < 384 * 4; i += 512) sWih[i] = W_ih[i];
    for (int i = tid; i < 384; i += 512) { sBih[i] = b_ih[i]; sBhh[i] = b_hh[i]; }
    for (int i = tid; i < B_DIM * HS; i += 512) sH[i] = (_Float16)0.0f;
    __syncthreads();

    // h in f32, C-layout: hc[q][j] = h(row rb+grp8+j, col co+q*16+l15)
    v8f hc[4] = {};
    float wsc[4];
#pragma unroll
    for (int q = 0; q < 4; ++q) wsc[q] = W_sc[co + q * 16 + l15];
    const float bsc = b_sc[0];

    int zoff = 0;                         // laundered 0: defeats LICM hoisting
#pragma unroll 1
    for (int t = 0; t < T_DIM; ++t) {
        asm volatile("" : "+v"(zoff));    // weight-fragment loads stay in-loop
        const _Float16* wp = &sW[0] + zoff;   // keeps addrspace(3) -> ds_load

        if (tid < B_DIM) {
            const float* xp = x + (size_t)tid * T_DIM * 4 + (size_t)t * 4;
#pragma unroll
            for (int d = 0; d < 4; ++d) sX[tid * 4 + d] = xp[d];
            if (t + 1 < T_DIM) __builtin_prefetch(xp + 4, 0, 3);  // global_prefetch_b8
        }
        __syncthreads();   // B1: sX staged; prior-step sH writes visible

        // gh(half) = h @ W_hh_s^T : 4 K-slices x 12 N-tiles (r|z|n half-cols)
        // double-buffered: while WMMA consumes (a,bcur), next frag is in flight
        v8f acc[12] = {};
        const _Float16* hbase = sH + rb * HS;
        v16h a    = load_frag_a(hbase, HS, lane, 0);
        v16h bcur = load_frag_b(wp, HS, co, 0, lane);       // q=0, kt=0
#pragma unroll
        for (int kt = 0; kt < 4; ++kt) {
            const int ktn = (kt < 3) ? kt + 1 : kt;
            v16h anext = load_frag_a(hbase, HS, lane, ktn * 32);
#pragma unroll
            for (int q = 0; q < 12; ++q) {
                const int qn  = (q < 11) ? q + 1 : 0;
                const int ktq = (q < 11) ? kt : ktn;
                const int n0n = (qn >> 2) * 128 + co + (qn & 3) * 16;
                v16h bnext = load_frag_b(wp, HS, n0n, ktq * 32, lane);
                acc[q] = WMMA_F16(a, bcur, acc[q]);
                bcur = bnext;
            }
            a = anext;
        }
        __syncthreads();   // B2: all reads of h(t-1) complete before updates

        // x rows for the 8 (m, j) slots of this lane
        float x4[8][4];
#pragma unroll
        for (int j = 0; j < 8; ++j) {
            int br = rb + grp8 + j;
#pragma unroll
            for (int d = 0; d < 4; ++d) x4[j][d] = sX[br * 4 + d];
        }

        float psc[8] = {};
#pragma unroll
        for (int q = 0; q < 4; ++q) {
            const int cr = co + q * 16 + l15;  // h col == r-gate col
            const int cz = 128 + cr;
            const int cn = 256 + cr;
            const float wr0 = sWih[cr * 4 + 0], wr1 = sWih[cr * 4 + 1],
                        wr2 = sWih[cr * 4 + 2], wr3 = sWih[cr * 4 + 3];
            const float wz0 = sWih[cz * 4 + 0], wz1 = sWih[cz * 4 + 1],
                        wz2 = sWih[cz * 4 + 2], wz3 = sWih[cz * 4 + 3];
            const float wn0 = sWih[cn * 4 + 0], wn1 = sWih[cn * 4 + 1],
                        wn2 = sWih[cn * 4 + 2], wn3 = sWih[cn * 4 + 3];
            const float bir = sBih[cr], biz = sBih[cz], bin_ = sBih[cn];
            const float bhr = sBhh[cr], bhz = sBhh[cz], bhn  = sBhh[cn];
#pragma unroll
            for (int j = 0; j < 8; ++j) {
                float gir = bir + x4[j][0]*wr0 + x4[j][1]*wr1 + x4[j][2]*wr2 + x4[j][3]*wr3;
                float giz = biz + x4[j][0]*wz0 + x4[j][1]*wz1 + x4[j][2]*wz2 + x4[j][3]*wz3;
                float gin = bin_ + x4[j][0]*wn0 + x4[j][1]*wn1 + x4[j][2]*wn2 + x4[j][3]*wn3;
                float hr = acc[q][j]     + bhr;
                float hz = acc[4 + q][j] + bhz;
                float hn = acc[8 + q][j] + bhn;
                float r = sigmf_(gir + hr);
                float z = sigmf_(giz + hz);
                float n = tanhf(gin + r * hn);
                float hnew = (1.0f - z) * n + z * hc[q][j];
                hc[q][j] = hnew;
                psc[j] += hnew * wsc[q];
                int br = rb + grp8 + j;
                _Float16 h16 = (_Float16)hnew;
                sH[br * HS + cr] = h16;
                slow16[((size_t)br * T_DIM + t) * HS + cr] = h16;
            }
        }

        // sc head: half-dot reduced across lane group, partial stored per half
        float scv = 0.0f;
#pragma unroll
        for (int j = 0; j < 8; ++j) {
            float s = red16_(psc[j]);
            if (l15 == j) scv = s;
        }
        if (l15 < 8) {
            int br = rb + grp8 + l15;
            sSC[br * 2 + p] = scv;
        }
        __syncthreads();   // B3: sSC complete; h(t) complete for next matmul
        if (p == 0 && l15 < 8) {
            int br = rb + grp8 + l15;
            sc_out[(size_t)br * T_DIM + t] =
                sigmf_(sSC[br * 2 + 0] + sSC[br * 2 + 1] + bsc);
        }
    }
}

// ---------------------------------------------------------------------------
// Fast GRU + heads: each wave owns a 16-row M-tile of the 65536-row batch.
// Projection h0 = slow @ W_proj^T + b_proj, then 15 GRU steps with zero input,
// evidence/logit heads, first-passage tau, rt/fc.
__global__ __launch_bounds__(256) void fast_gru_kernel(
    const _Float16* __restrict__ slow16,  // (N,128) f16
    const float* __restrict__ W_proj,     // (64,128)
    const float* __restrict__ b_proj,     // (64,)
    const float* __restrict__ W_hh_f,     // (192,64)
    const float* __restrict__ b_ih_f,     // (192,)
    const float* __restrict__ b_hh_f,     // (192,)
    const float* __restrict__ w_init,     // (1,64)
    const float* __restrict__ w_ev,       // (1,64)
    const float* __restrict__ W_fc,       // (1,64)
    const float* __restrict__ b_fc,       // (1,)
    const float* __restrict__ theta_p,    // scalar
    float* __restrict__ out)              // full d_out
{
    __shared__ _Float16 sWp[HF * HS];       // 16 KB: W_proj f16, [n][k]
    __shared__ _Float16 sWf[3 * HF * HF];   // 24 KB: W_hh_f f16, [n][k]
    __shared__ _Float16 sHt[8][16 * HF];    // 16 KB: per-wave h transpose scratch

    const int tid  = threadIdx.x;
    const int lane = tid & 31;
    const int wave = tid >> 5;
    const int l15  = lane & 15;
    const int grp8 = (lane & 16) ? 8 : 0;

    for (int i = tid; i < HF * HS; i += 256)     sWp[i] = (_Float16)W_proj[i];
    for (int i = tid; i < 3 * HF * HF; i += 256) sWf[i] = (_Float16)W_hh_f[i];
    __syncthreads();

    const size_t R = ((size_t)blockIdx.x * 8 + wave) * 16;  // first row of M-tile

    // per-lane broadcast constants (indexed by this lane's column slot)
    float bp4[4], wev4[4], wi4[4], wfc4[4], bihf[12], bhhf[12];
#pragma unroll
    for (int t = 0; t < 4; ++t) {
        int c = t * 16 + l15;
        bp4[t] = b_proj[c]; wev4[t] = w_ev[c]; wi4[t] = w_init[c]; wfc4[t] = W_fc[c];
    }
#pragma unroll
    for (int t = 0; t < 12; ++t) {
        int c = t * 16 + l15;
        bihf[t] = b_ih_f[c]; bhhf[t] = b_hh_f[c];
    }
    const float theta = theta_p[0];
    const float bfc   = b_fc[0];

    // ---- projection: (16x128) @ (128x64), double-buffered B ----
    v8f hc4[4] = {};
    {
        v16h a    = load_frag_a(slow16 + R * HS, HS, lane, 0);
        v16h bcur = load_frag_b(sWp, HS, 0, 0, lane);
#pragma unroll
        for (int kt = 0; kt < 4; ++kt) {
            const int ktn = (kt < 3) ? kt + 1 : kt;
            v16h anext = load_frag_a(slow16 + R * HS, HS, lane, ktn * 32);
#pragma unroll
            for (int nt = 0; nt < 4; ++nt) {
                const int nn  = (nt < 3) ? nt + 1 : 0;
                const int ktq = (nt < 3) ? kt : ktn;
                v16h bnext = load_frag_b(sWp, HS, nn * 16, ktq * 32, lane);
                hc4[nt] = WMMA_F16(a, bcur, hc4[nt]);
                bcur = bnext;
            }
            a = anext;
        }
    }
    _Float16* myH = &sHt[wave][0];
#pragma unroll
    for (int nt = 0; nt < 4; ++nt) {
#pragma unroll
        for (int j = 0; j < 8; ++j) {
            hc4[nt][j] += bp4[nt];
            myH[(grp8 + j) * HF + nt * 16 + l15] = (_Float16)hc4[nt][j];
        }
    }

    float tauf = 15.0f, sel_logit = 0.0f;
    float* out_rt  = out;
    float* out_fc  = out + 2 * (size_t)NBT;
    float* out_phi = out + 3 * (size_t)NBT;
    float* out_lg  = out + 3 * (size_t)NBT + (size_t)NBT * KSTEPS;

    int zoff = 0;                         // laundered 0: defeats LICM hoisting
#pragma unroll 1
    for (int k = 0; k < KSTEPS; ++k) {
        asm volatile("" : "+v"(zoff));
        const _Float16* wf = &sWf[0] + zoff;

        // same-wave LDS RAW: stores above must land before A-fragment reads
        asm volatile("s_wait_dscnt 0" ::: "memory");
        v16h a0 = load_frag_a(myH, HF, lane, 0);
        v16h a1 = load_frag_a(myH, HF, lane, 32);

        // 24 WMMAs (2 K-slices x 12 N-tiles), double-buffered B fragments
        v8f acc[12] = {};
        v16h bcur = load_frag_b(wf, HF, 0, 0, lane);
#pragma unroll
        for (int i = 0; i < 24; ++i) {
            const int nt = (i < 12) ? i : i - 12;
            const int in = (i < 23) ? i + 1 : 23;
            const int ntn = (in < 12) ? in : in - 12;
            const int k0n = (in < 12) ? 0 : 32;
            v16h bnext = load_frag_b(wf, HF, ntn * 16, k0n, lane);
            acc[nt] = WMMA_F16((i < 12) ? a0 : a1, bcur, acc[nt]);
            bcur = bnext;
        }

        float pphi[8] = {}, plog[8] = {};
#pragma unroll
        for (int ct = 0; ct < 4; ++ct) {
            float we = wev4[ct] + ((k == 0) ? wi4[ct] : 0.0f);  // phi0 = phi + h.w_init
#pragma unroll
            for (int j = 0; j < 8; ++j) {
                float hr = acc[ct][j]     + bhhf[ct];
                float hz = acc[4 + ct][j] + bhhf[4 + ct];
                float hn = acc[8 + ct][j] + bhhf[8 + ct];
                float r = sigmf_(bihf[ct] + hr);
                float z = sigmf_(bihf[4 + ct] + hz);
                float n = tanhf(bihf[8 + ct] + r * hn);
                float hnew = (1.0f - z) * n + z * hc4[ct][j];
                hc4[ct][j] = hnew;
                pphi[j] += hnew * we;
                plog[j] += hnew * wfc4[ct];
                myH[(grp8 + j) * HF + ct * 16 + l15] = (_Float16)hnew;
            }
        }

        float phiv = 0.0f, logv = 0.0f;
#pragma unroll
        for (int j = 0; j < 8; ++j) {
            float sp = red16_(pphi[j]);
            float sl = red16_(plog[j]);
            if (l15 == j) { phiv = sp; logv = sl; }
        }
        if (l15 < 8) {
            size_t n_ = R + grp8 + l15;
            logv += bfc;
            out_phi[n_ * KSTEPS + k] = phiv;
            out_lg[n_ * KSTEPS + k]  = logv;
            if (tauf > 14.5f) {                 // not yet crossed
                sel_logit = logv;               // tracks logits[min(tau, K-1)]
                if (phiv > theta) tauf = (float)k;
            }
        }
    }

    if (l15 < 8) {
        size_t n_ = R + grp8 + l15;
        out_rt[n_] = tauf * DELTA_T;
        out_fc[n_] = sigmf_(sel_logit);
    }
}

// ---------------------------------------------------------------------------
extern "C" void kernel_launch(void* const* d_in, const int* in_sizes, int n_in,
                              void* d_out, int out_size, void* d_ws, size_t ws_size,
                              hipStream_t stream) {
    (void)in_sizes; (void)n_in; (void)out_size; (void)ws_size;

    const float* x      = (const float*)d_in[0];
    const float* W_ih_s = (const float*)d_in[1];
    const float* W_hh_s = (const float*)d_in[2];
    const float* b_ih_s = (const float*)d_in[3];
    const float* b_hh_s = (const float*)d_in[4];
    const float* W_proj = (const float*)d_in[5];
    const float* b_proj = (const float*)d_in[6];
    const float* W_hh_f = (const float*)d_in[7];
    const float* b_ih_f = (const float*)d_in[8];
    const float* b_hh_f = (const float*)d_in[9];
    const float* w_init = (const float*)d_in[10];
    const float* w_ev   = (const float*)d_in[11];
    const float* W_fc   = (const float*)d_in[12];
    const float* b_fc   = (const float*)d_in[13];
    const float* W_sc   = (const float*)d_in[14];
    const float* b_sc   = (const float*)d_in[15];
    const float* theta  = (const float*)d_in[16];

    float* out = (float*)d_out;

    // workspace: [0, 16MB) slow_out f16
    _Float16* slow16 = (_Float16*)d_ws;

    slow_gru_kernel<<<1, 512, 0, stream>>>(x, W_hh_s, W_ih_s, b_ih_s, b_hh_s,
                                           W_sc, b_sc, slow16, out + (size_t)NBT);

    fast_gru_kernel<<<NBT / 128, 256, 0, stream>>>(slow16, W_proj, b_proj, W_hh_f,
                                                   b_ih_f, b_hh_f, w_init, w_ev,
                                                   W_fc, b_fc, theta, out);
}
